// CRF_2765958938822
// MI455X (gfx1250) — matrix operand order repaired and checked by reading, not compiled, not run on hardware
//
#include <hip/hip_runtime.h>
#include <hip/hip_bf16.h>
#include <math.h>

typedef _Float16 v16h __attribute__((ext_vector_type(16)));
typedef _Float16 v8h  __attribute__((ext_vector_type(8)));
typedef float    v8f  __attribute__((ext_vector_type(8)));

#define BB 512
#define TT 512
#define KK 128

#define LOG2E 1.4426950408889634f
#define LN2   0.6931471805599453f

// ---------------------------------------------------------------------------
// Forward scan: one workgroup = 16 batch rows, full T-scan, WMMA log-matmul.
// Scores kept in log2 domain (beta = alpha*log2e): phase 2/4 are raw
// v_exp_f32 / v_log_f32 with no range-fixup code.
// ---------------------------------------------------------------------------
__global__ __launch_bounds__(256) void crf_forward_kernel(
    const float* __restrict__ em,          // (B,T,K)
    const unsigned char* __restrict__ mask,// (B,T)
    const float* __restrict__ trans,       // (K,K)  trans[i*K+j] = from i to j
    const float* __restrict__ startT,      // (K)
    const float* __restrict__ endT,        // (K)
    float* __restrict__ log_den)           // (B)
{
    // Et[j][i] = exp(trans[i][j]) = 2^(trans*log2e)  (transposed for B-frags)
    __shared__ __align__(32) _Float16 Et[KK][KK];    // 32 KB
    __shared__ __align__(16) float    beta[16][KK];  //  8 KB  log2-domain alpha
    __shared__ __align__(16) _Float16 p[16][KK];     //  4 KB  2^(beta - m)
    __shared__ __align__(16) float    mbuf[2][16];   // double-buffered row max

    const int tid  = threadIdx.x;
    const int w    = tid >> 5;     // wave id 0..7  -> owns output cols 16w..16w+15
    const int l    = tid & 31;     // lane in wave32
    const int h    = l >> 4;       // lane half
    const int ln   = l & 15;
    const int bb   = blockIdx.x * 16;     // batch tile base
    const int jcol = w * 16 + ln;         // this lane's output column (B/C/D layout)

    // --- E = exp(trans), transposed, f16 ---
    for (int idx = tid; idx < KK * KK; idx += 256) {
        int i = idx >> 7, j = idx & (KK - 1);
        Et[j][i] = (_Float16)__builtin_amdgcn_exp2f(trans[idx] * LOG2E);
    }
    // --- beta0 = (start + emissions[:,0,:]) * log2e ---
    for (int idx = tid; idx < 16 * KK; idx += 256) {
        int r = idx >> 7, j = idx & (KK - 1);
        beta[r][j] = (startT[j] + em[((size_t)(bb + r) * TT) * KK + j]) * LOG2E;
    }
    __syncthreads();

    // --- load B-fragments once (loop-invariant): B[k][n] = Et[jcol][k] ---
    // contiguous k order used here for B (same as A, not AMD's table order): lane = column n, K = 16*h + e  (e = half idx)
    v16h bf[4];
#pragma unroll
    for (int kb = 0; kb < 4; ++kb)
        bf[kb] = *(const v16h*)&Et[jcol][kb * 32 + h * 16];

    // --- per-lane emission/mask base pointers for phase 4 (row = r + 8h) ---
    const float*         ep[8];
    const unsigned char* mp[8];
    float                areg[8];   // this lane's own beta cells (row r+8h, col jcol)
#pragma unroll
    for (int r = 0; r < 8; ++r) {
        size_t b = (size_t)(bb + r + 8 * h);
        ep[r]   = em + b * TT * KK + jcol;
        mp[r]   = mask + b * TT;
        areg[r] = beta[r + 8 * h][jcol];
    }

    // --- software pipeline: preload emissions/mask for t = 1 ---
    float         emv[8];
    unsigned char mkv[8];
#pragma unroll
    for (int r = 0; r < 8; ++r) {
        emv[r] = ep[r][(size_t)1 * KK];
        mkv[r] = mp[r][1];
    }

    // --- initial row max of beta0 into mbuf[1] (used at t=1) ---
    {
        int row = 2 * w + h;
        float mx = -INFINITY;
#pragma unroll
        for (int c = 0; c < KK; c += 16) mx = fmaxf(mx, beta[row][c + ln]);
        mx = fmaxf(mx, __shfl_xor(mx, 8, 32));
        mx = fmaxf(mx, __shfl_xor(mx, 4, 32));
        mx = fmaxf(mx, __shfl_xor(mx, 2, 32));
        mx = fmaxf(mx, __shfl_xor(mx, 1, 32));
        if (ln == 0) mbuf[1][row] = mx;
    }
    __syncthreads();

    // --------------------------- time scan (2 barriers/step) ----------------
    for (int t = 1; t < TT; ++t) {
        const int buf  = t & 1;
        const int nbuf = buf ^ 1;

        // issue next step's emission/mask loads NOW (consumed next iteration)
        const int tn = (t + 1 < TT) ? t + 1 : t;
        float         emn[8];
        unsigned char mkn[8];
#pragma unroll
        for (int r = 0; r < 8; ++r) {
            emn[r] = ep[r][(size_t)tn * KK];
            mkn[r] = mp[r][tn];
        }

        // phase 2: p = 2^(beta - m) f16; each thread owns 8 CONTIGUOUS columns
        // (vectorized: 2x ds_load_b128 in, 1x ds_store_b128 out).
        // Also reset the other max buffer for this step's phase-4 atomics.
        {
            int row = tid >> 4;
            int c0  = (tid & 15) * 8;
            float m = mbuf[buf][row];
            if (tid < 16) mbuf[nbuf][tid] = -INFINITY;
            float4 b0 = *(const float4*)&beta[row][c0];
            float4 b1 = *(const float4*)&beta[row][c0 + 4];
            v8h ph;
            ph[0] = (_Float16)__builtin_amdgcn_exp2f(b0.x - m);
            ph[1] = (_Float16)__builtin_amdgcn_exp2f(b0.y - m);
            ph[2] = (_Float16)__builtin_amdgcn_exp2f(b0.z - m);
            ph[3] = (_Float16)__builtin_amdgcn_exp2f(b0.w - m);
            ph[4] = (_Float16)__builtin_amdgcn_exp2f(b1.x - m);
            ph[5] = (_Float16)__builtin_amdgcn_exp2f(b1.y - m);
            ph[6] = (_Float16)__builtin_amdgcn_exp2f(b1.z - m);
            ph[7] = (_Float16)__builtin_amdgcn_exp2f(b1.w - m);
            *(v8h*)&p[row][c0] = ph;
        }
        __syncthreads();

        // phase 3: acc = p(16x128) @ E(128x16); all A-loads first, 2 WMMA chains
        // A k order used here (contiguous, matches B): row M = ln, K = 16*h + e
        v16h af[4];
#pragma unroll
        for (int kb = 0; kb < 4; ++kb) {
            v8h lo = *(const v8h*)&p[ln][kb * 32 + h * 16];       // K = 16h..16h+7
            v8h hi = *(const v8h*)&p[ln][kb * 32 + h * 16 + 8];   // K = 16h+8..
            af[kb] = __builtin_shufflevector(lo, hi,
                0, 1, 2, 3, 4, 5, 6, 7, 8, 9, 10, 11, 12, 13, 14, 15);
        }
        v8f acc0 = {}, acc1 = {};   // two independent chains, depth 2 each
        acc0 = __builtin_amdgcn_wmma_f32_16x16x32_f16(
            false, af[0], false, bf[0], (short)0, acc0, false, false);
        acc1 = __builtin_amdgcn_wmma_f32_16x16x32_f16(
            false, af[2], false, bf[2], (short)0, acc1, false, false);
        acc0 = __builtin_amdgcn_wmma_f32_16x16x32_f16(
            false, af[1], false, bf[1], (short)0, acc0, false, false);
        acc1 = __builtin_amdgcn_wmma_f32_16x16x32_f16(
            false, af[3], false, bf[3], (short)0, acc1, false, false);

        // per-row maxes for my 8 rows: contiguous -> two 16B LDS loads
        float4 mlo = *(const float4*)&mbuf[buf][8 * h + 0];
        float4 mhi = *(const float4*)&mbuf[buf][8 * h + 4];
        float  mv[8] = { mlo.x, mlo.y, mlo.z, mlo.w, mhi.x, mhi.y, mhi.z, mhi.w };

        // phase 4: beta' = log2(acc) + m + em*log2e; BRANCHLESS mask blend;
        // fold row-max accumulation via LDS float atomic max (ds_max_num_f32).
#pragma unroll
        for (int r = 0; r < 8; ++r) {
            int orow = r + 8 * h;
            float s      = acc0[r] + acc1[r];
            float newv   = __builtin_amdgcn_logf(s) + fmaf(emv[r], LOG2E, mv[r]);
            float mf     = (float)mkv[r];                    // 0.0f or 1.0f
            float chosen = fmaf(mf, newv - areg[r], areg[r]);
            areg[r] = chosen;
            beta[orow][jcol] = chosen;
            __hip_atomic_fetch_max(&mbuf[nbuf][orow], chosen,
                                   __ATOMIC_RELAXED, __HIP_MEMORY_SCOPE_WORKGROUP);
            emv[r] = emn[r];
            mkv[r] = mkn[r];
        }
        __syncthreads();
    }

    // --- log_den[b] = ln-domain logsumexp_j(beta[b][j] + end[j]*log2e) * ln2 ---
    {
        int row = 2 * w + h;
        float v[8];
        float mx = -INFINITY;
#pragma unroll
        for (int u = 0; u < 8; ++u) {
            int c = u * 16 + ln;
            v[u] = fmaf(endT[c], LOG2E, beta[row][c]);
            mx = fmaxf(mx, v[u]);
        }
        mx = fmaxf(mx, __shfl_xor(mx, 8, 32));
        mx = fmaxf(mx, __shfl_xor(mx, 4, 32));
        mx = fmaxf(mx, __shfl_xor(mx, 2, 32));
        mx = fmaxf(mx, __shfl_xor(mx, 1, 32));
        float s = 0.f;
#pragma unroll
        for (int u = 0; u < 8; ++u) s += __builtin_amdgcn_exp2f(v[u] - mx);
        s += __shfl_xor(s, 8, 32);
        s += __shfl_xor(s, 4, 32);
        s += __shfl_xor(s, 2, 32);
        s += __shfl_xor(s, 1, 32);
        if (ln == 0)
            log_den[bb + row] = (__builtin_amdgcn_logf(s) + mx) * LN2;
    }
}

// ---------------------------------------------------------------------------
// Gold (numerator) score: one thread per batch row.
// ---------------------------------------------------------------------------
__global__ __launch_bounds__(256) void crf_gold_kernel(
    const float* __restrict__ em, const long long* __restrict__ tags,
    const unsigned char* __restrict__ mask, const float* __restrict__ trans,
    const float* __restrict__ startT, const float* __restrict__ endT,
    float* __restrict__ log_num)
{
    int b = blockIdx.x * blockDim.x + threadIdx.x;
    if (b >= BB) return;
    const long long*     tg = tags + (size_t)b * TT;
    const unsigned char* mk = mask + (size_t)b * TT;
    const float*         eb = em + (size_t)b * TT * KK;

    int   prev  = (int)tg[0];
    float score = startT[prev];
    int   cnt   = 0;
    for (int t = 0; t < TT; ++t) {
        int   cur = (int)tg[t];
        float mf  = mk[t] ? 1.f : 0.f;
        score += eb[(size_t)t * KK + cur] * mf;
        if (t > 0) score += trans[cur * KK + prev] * mf;  // trans[tags[t], tags[t-1]]
        cnt  += mk[t] ? 1 : 0;
        prev = cur;
    }
    int lastIdx = cnt > 0 ? cnt - 1 : 0;
    score += endT[(int)tg[lastIdx]];
    log_num[b] = score;
}

// ---------------------------------------------------------------------------
// Deterministic tree reduction: out = -mean(log_num - log_den)
// ---------------------------------------------------------------------------
__global__ __launch_bounds__(256) void crf_reduce_kernel(
    const float* __restrict__ log_den, const float* __restrict__ log_num,
    float* __restrict__ out)
{
    __shared__ float sm[8];
    int tid = threadIdx.x;
    float s = 0.f;
    for (int i = tid; i < BB; i += 256) s += log_num[i] - log_den[i];
    s += __shfl_xor(s, 16, 32);
    s += __shfl_xor(s, 8, 32);
    s += __shfl_xor(s, 4, 32);
    s += __shfl_xor(s, 2, 32);
    s += __shfl_xor(s, 1, 32);
    if ((tid & 31) == 0) sm[tid >> 5] = s;
    __syncthreads();
    if (tid == 0) {
        float tot = 0.f;
#pragma unroll
        for (int i = 0; i < 8; ++i) tot += sm[i];
        out[0] = -tot / (float)BB;
    }
}

extern "C" void kernel_launch(void* const* d_in, const int* in_sizes, int n_in,
                              void* d_out, int out_size, void* d_ws, size_t ws_size,
                              hipStream_t stream) {
    const float*         em    = (const float*)d_in[0];
    const long long*     tags  = (const long long*)d_in[1];
    const unsigned char* mask  = (const unsigned char*)d_in[2];
    const float*         trans = (const float*)d_in[3];
    const float*         stT   = (const float*)d_in[4];
    const float*         enT   = (const float*)d_in[5];

    float* wsf     = (float*)d_ws;
    float* log_den = wsf;        // 512 floats
    float* log_num = wsf + BB;   // 512 floats

    crf_forward_kernel<<<dim3(BB / 16), dim3(256), 0, stream>>>(
        em, mask, trans, stT, enT, log_den);
    crf_gold_kernel<<<dim3(2), dim3(256), 0, stream>>>(
        em, tags, mask, trans, stT, enT, log_num);
    crf_reduce_kernel<<<dim3(1), dim3(256), 0, stream>>>(
        log_den, log_num, (float*)d_out);
}
